// GNN_43293270344244
// MI455X (gfx1250) — matrix operand order repaired and checked by reading, not compiled, not run on hardware
//
#include <hip/hip_runtime.h>
#include <math.h>

#define N_NODES 10000
#define N_EDGES 160000
#define WIDTH   256
#define HID     512
#define EW      128
#define HEAD_DIM 64
#define NHEAD    8

typedef __bf16 bf16_t;
typedef __attribute__((ext_vector_type(16))) __bf16 v16bf;
typedef __attribute__((ext_vector_type(8)))  float  v8f;

// ---- WMMA fragment loaders (CDNA5 ISA 7.12.2 layouts, wave32) ----
// A (16x32 bf16, row-major source in LDS): lane<16 holds K0-7 & K16-23 of row
// m=lane; lane>=16 holds K8-15 & K24-31 of row m=lane-16.
__device__ __forceinline__ v16bf load_a_frag(const bf16_t* A, int lda, int lane) {
  const int m  = lane & 15;
  const int kq = (lane >> 4) << 3;          // 0 or 8
  const bf16_t* p = A + m * lda + kq;
  v16bf a;
#pragma unroll
  for (int j = 0; j < 8; ++j) { a[j] = p[j]; a[8 + j] = p[16 + j]; }
  return a;
}

// B fragments are pre-packed (fragment-major): tile (kt,ct) occupies 512
// contiguous bf16, lane's 16 values contiguous -> one aligned 32B vector load.
__device__ __forceinline__ v16bf load_b_packed(const bf16_t* __restrict__ P,
                                               int kt, int ct, int numCT, int lane) {
  return *(const v16bf*)(P + (((size_t)(kt * numCT + ct)) << 9) + (lane << 4));
}

__device__ __forceinline__ float gelu_erf(float x) {
  return 0.5f * x * (1.0f + erff(x * 0.70710678118654752f));
}

// ---- pack: W (KxN row-major f32) -> bf16 fragment-major ----
__global__ void k_pack_b(const float* __restrict__ W, bf16_t* __restrict__ P,
                         int K, int Ncols) {
  int idx = blockIdx.x * blockDim.x + threadIdx.x;
  if (idx >= K * Ncols) return;
  int tile = idx >> 9;              // /512 elements per 32x16 tile
  int within = idx & 511;
  int lane = within >> 4, j = within & 15;
  int numCT = Ncols >> 4;
  int kt = tile / numCT, ct = tile - kt * numCT;
  int n = lane & 15, kb = (lane >> 4) << 3;
  P[idx] = (bf16_t)W[(kt * 32 + kb + (j & 7) + ((j >> 3) << 4)) * Ncols + ct * 16 + n];
}

__global__ void k_zero(float* __restrict__ p, int n) {
  int i = blockIdx.x * blockDim.x + threadIdx.x;
  if (i < n) p[i] = 0.0f;
}

// ---- pre: xx = LN(x @ W_pre + b_pre); store bf16 xxh ----
__global__ __launch_bounds__(512) void k_pre(const float* __restrict__ x,
                                             const bf16_t* __restrict__ Pw,
                                             const float* __restrict__ b,
                                             bf16_t* __restrict__ xxh) {
  __shared__ bf16_t sA[16][WIDTH + 8];
  __shared__ float  sC[16][WIDTH];
  const int tid = threadIdx.x, wave = tid >> 5, lane = tid & 31;
  const int n0 = blockIdx.x * 16;
  for (int i = tid; i < 16 * WIDTH; i += 512) {
    int r = i >> 8, c = i & (WIDTH - 1);
    sA[r][c] = (bf16_t)x[(n0 + r) * WIDTH + c];
  }
  __syncthreads();
  const int ct = wave;                       // 16 waves cover 256 cols
  v8f acc = {};
#pragma unroll
  for (int kt = 0; kt < WIDTH / 32; ++kt) {
    v16bf a  = load_a_frag(&sA[0][kt * 32], WIDTH + 8, lane);
    v16bf bb = load_b_packed(Pw, kt, ct, WIDTH / 16, lane);
    acc = __builtin_amdgcn_wmma_f32_16x16x32_bf16(false, a, false, bb, (short)0, acc, false, false);
  }
  const int cn = ct * 16 + (lane & 15);
  const float bias = b[cn];
  const int mb = (lane >> 4) << 3;
#pragma unroll
  for (int r = 0; r < 8; ++r) sC[mb + r][cn] = acc[r] + bias;
  __syncthreads();
  { // LayerNorm: wave w handles row w
    const int r = wave;
    float s = 0.0f;
#pragma unroll
    for (int j = 0; j < 8; ++j) s += sC[r][lane + 32 * j];
#pragma unroll
    for (int o = 16; o > 0; o >>= 1) s += __shfl_down(s, o, 32);
    const float mean = __shfl(s, 0, 32) * (1.0f / WIDTH);
    float v = 0.0f;
#pragma unroll
    for (int j = 0; j < 8; ++j) { float d = sC[r][lane + 32 * j] - mean; v += d * d; }
#pragma unroll
    for (int o = 16; o > 0; o >>= 1) v += __shfl_down(v, o, 32);
    const float rinv = rsqrtf(__shfl(v, 0, 32) * (1.0f / WIDTH) + 1e-5f);
#pragma unroll
    for (int j = 0; j < 8; ++j) {
      int c = lane + 32 * j;
      xxh[(n0 + r) * WIDTH + c] = (bf16_t)((sC[r][c] - mean) * rinv);
    }
  }
}

// ---- fused node GEMMs: qn/kn/vn = xx@W+b, m0 = gelu(xx@W_msg0+b) ----
__global__ __launch_bounds__(256) void k_node4(
    const bf16_t* __restrict__ xxh,
    const bf16_t* __restrict__ Pq, const bf16_t* __restrict__ Pk,
    const bf16_t* __restrict__ Pv, const bf16_t* __restrict__ Pm,
    const float* __restrict__ bq, const float* __restrict__ bk,
    const float* __restrict__ bv, const float* __restrict__ bm,
    float* __restrict__ qn, float* __restrict__ kn,
    float* __restrict__ vn, float* __restrict__ m0) {
  __shared__ bf16_t sA[16][WIDTH + 8];
  const int tid = threadIdx.x, wave = tid >> 5, lane = tid & 31;
  const int n0 = blockIdx.x * 16;
  for (int i = tid; i < 16 * WIDTH; i += 256) {
    int r = i >> 8, c = i & (WIDTH - 1);
    sA[r][c] = xxh[(n0 + r) * WIDTH + c];
  }
  __syncthreads();
  v16bf afr[WIDTH / 32];                       // 8 frags, reused 16x
#pragma unroll
  for (int kt = 0; kt < WIDTH / 32; ++kt)
    afr[kt] = load_a_frag(&sA[0][kt * 32], WIDTH + 8, lane);

  const bf16_t* Ps[4]  = {Pq, Pk, Pv, Pm};
  const float*  bs[4]  = {bq, bk, bv, bm};
  float*        os[4]  = {qn, kn, vn, m0};
  const int mb = (lane >> 4) << 3;
#pragma unroll
  for (int w = 0; w < 4; ++w) {
    for (int ct = wave; ct < HID / 16; ct += 8) {
      v8f acc = {};
#pragma unroll
      for (int kt = 0; kt < WIDTH / 32; ++kt) {
        v16bf bb = load_b_packed(Ps[w], kt, ct, HID / 16, lane);
        acc = __builtin_amdgcn_wmma_f32_16x16x32_bf16(false, afr[kt], false, bb, (short)0, acc, false, false);
      }
      const int cn = ct * 16 + (lane & 15);
      const float bias = bs[w][cn];
#pragma unroll
      for (int r = 0; r < 8; ++r) {
        float v = acc[r] + bias;
        if (w == 3) v = gelu_erf(v);
        os[w][(n0 + mb + r) * HID + cn] = v;
      }
    }
  }
}

// ---- edge kernel: ee build, ee@W_r{0,1,2}, per-head attention, scatter-add ----
__global__ __launch_bounds__(256) void k_edge(
    const int* __restrict__ edge_index, const int* __restrict__ edge_attr,
    const float* __restrict__ edge_embed,
    const float* __restrict__ emb0, const float* __restrict__ emb1,
    const float* __restrict__ emb2, const float* __restrict__ emb3,
    const float* __restrict__ init0_e, const float* __restrict__ init0,
    const bf16_t* __restrict__ Pr0, const bf16_t* __restrict__ Pr1, const bf16_t* __restrict__ Pr2,
    const float* __restrict__ br0, const float* __restrict__ br1, const float* __restrict__ br2,
    const float* __restrict__ qn, const float* __restrict__ kn, const float* __restrict__ vn,
    float* __restrict__ agg) {
  __shared__ bf16_t sE[16][EW + 8];
  __shared__ float  sDot[16][NHEAD];
  __shared__ int ssrc[16], sdst[16], sattr[16][4];
  const int tid = threadIdx.x, wave = tid >> 5, lane = tid & 31;
  const int e0 = blockIdx.x * 16;

  if (tid < 16) { ssrc[tid] = edge_index[e0 + tid]; sdst[tid] = edge_index[N_EDGES + e0 + tid]; }
  if (tid >= 64 && tid < 128) {
    int t = tid - 64;
    sattr[t >> 2][t & 3] = edge_attr[(e0 + (t >> 2)) * 4 + (t & 3)];
  }
  if (tid < 16 * NHEAD) ((float*)sDot)[tid] = 0.0f;
  float w0 = __expf(init0_e[0]), w1 = __expf(init0_e[1]),
        w2 = __expf(init0_e[2]), w3 = __expf(init0_e[3]);
  const float rs = rsqrtf(w0 + w1 + w2 + w3);
  w0 *= rs; w1 *= rs; w2 *= rs; w3 *= rs;
  const float c0 = init0[0], c1 = init0[1];
  const float s_e = __expf(init0[2]), s_v = __expf(init0[3]);
  __syncthreads();

  for (int i = tid; i < 16 * EW; i += 256) {
    int e = i >> 7, c = i & (EW - 1);
    float v = emb0[sattr[e][0] * EW + c] * w0 + emb1[sattr[e][1] * EW + c] * w1
            + emb2[sattr[e][2] * EW + c] * w2 + emb3[sattr[e][3] * EW + c] * w3
            + edge_embed[(e0 + e) * EW + c];
    sE[e][c] = (bf16_t)(0.5f * v);
  }
  __syncthreads();

  v16bf afr[EW / 32];                         // 4 frags, reused by q,k,v GEMMs
#pragma unroll
  for (int kt = 0; kt < EW / 32; ++kt)
    afr[kt] = load_a_frag(&sE[0][kt * 32], EW + 8, lane);

  const int mb = (lane >> 4) << 3;
  // Phase 1: q & k tiles; in-register q.k head reduction (16-col tile lies in one head)
  for (int ct = wave; ct < HID / 16; ct += 8) {
    v8f accq = {}, acck = {};
#pragma unroll
    for (int kt = 0; kt < EW / 32; ++kt) {
      v16bf b0 = load_b_packed(Pr0, kt, ct, HID / 16, lane);
      accq = __builtin_amdgcn_wmma_f32_16x16x32_bf16(false, afr[kt], false, b0, (short)0, accq, false, false);
      v16bf b1 = load_b_packed(Pr1, kt, ct, HID / 16, lane);
      acck = __builtin_amdgcn_wmma_f32_16x16x32_bf16(false, afr[kt], false, b1, (short)0, acck, false, false);
    }
    const int cn = ct * 16 + (lane & 15);
    const float bq = br0[cn], bk = br1[cn];
    float part[8];
#pragma unroll
    for (int r = 0; r < 8; ++r) {
      const int e = mb + r;
      const float qv = qn[sdst[e] * HID + cn] + (accq[r] + bq) * s_e;
      const float kv = kn[ssrc[e] * HID + cn] + (acck[r] + bk) * s_e;
      part[r] = qv * kv;
    }
#pragma unroll
    for (int m = 1; m < 16; m <<= 1)
#pragma unroll
      for (int r = 0; r < 8; ++r) part[r] += __shfl_xor(part[r], m, 32);
    if ((lane & 15) == 0) {
      const int h = ct >> 2;                  // ct*16/64
#pragma unroll
      for (int r = 0; r < 8; ++r) atomicAdd(&sDot[mb + r][h], part[r]);
    }
  }
  __syncthreads();
  if (tid < 16 * NHEAD) {
    const int e = tid >> 3, h = tid & 7;
    sDot[e][h] = __expf(sDot[e][h] * 0.125f * c0 + c1);   // att
  }
  __syncthreads();

  // Phase 2: v tile -> gelu -> *att -> scatter-add to agg[dst]
  for (int ct = wave; ct < HID / 16; ct += 8) {
    v8f acc = {};
#pragma unroll
    for (int kt = 0; kt < EW / 32; ++kt) {
      v16bf b2 = load_b_packed(Pr2, kt, ct, HID / 16, lane);
      acc = __builtin_amdgcn_wmma_f32_16x16x32_bf16(false, afr[kt], false, b2, (short)0, acc, false, false);
    }
    const int cn = ct * 16 + (lane & 15);
    const float bv = br2[cn];
    const int h = ct >> 2;
#pragma unroll
    for (int r = 0; r < 8; ++r) {
      const int e = mb + r;
      const float v = gelu_erf(vn[ssrc[e] * HID + cn] + (acc[r] + bv) * s_v);
      atomicAdd(&agg[sdst[e] * HID + cn], v * sDot[e][h]);
    }
  }
}

// ---- final: out = x + (m0 + agg) @ W_post + b_post ----
__global__ __launch_bounds__(256) void k_final(const float* __restrict__ m0,
                                               const float* __restrict__ agg,
                                               const bf16_t* __restrict__ Pp,
                                               const float* __restrict__ b,
                                               const float* __restrict__ x,
                                               float* __restrict__ out) {
  __shared__ bf16_t sA[16][HID + 8];
  const int tid = threadIdx.x, wave = tid >> 5, lane = tid & 31;
  const int n0 = blockIdx.x * 16;
  for (int i = tid; i < 16 * HID; i += 256) {
    int r = i >> 9, c = i & (HID - 1);
    int g = (n0 + r) * HID + c;
    sA[r][c] = (bf16_t)(m0[g] + agg[g]);
  }
  __syncthreads();
  const int mb = (lane >> 4) << 3;
  for (int ct = wave; ct < WIDTH / 16; ct += 8) {
    v8f acc = {};
#pragma unroll
    for (int kt = 0; kt < HID / 32; ++kt) {
      v16bf a  = load_a_frag(&sA[0][kt * 32], HID + 8, lane);
      v16bf bb = load_b_packed(Pp, kt, ct, WIDTH / 16, lane);
      acc = __builtin_amdgcn_wmma_f32_16x16x32_bf16(false, a, false, bb, (short)0, acc, false, false);
    }
    const int cn = ct * 16 + (lane & 15);
    const float bias = b[cn];
#pragma unroll
    for (int r = 0; r < 8; ++r) {
      int g = (n0 + mb + r) * WIDTH + cn;
      out[g] = x[g] + acc[r] + bias;
    }
  }
}

extern "C" void kernel_launch(void* const* d_in, const int* in_sizes, int n_in,
                              void* d_out, int out_size, void* d_ws, size_t ws_size,
                              hipStream_t stream) {
  (void)in_sizes; (void)n_in; (void)out_size; (void)ws_size;
  const float* x          = (const float*)d_in[0];
  const int*   edge_index = (const int*)d_in[1];
  const int*   edge_attr  = (const int*)d_in[2];
  const float* edge_embed = (const float*)d_in[3];
  const float* emb0 = (const float*)d_in[4];
  const float* emb1 = (const float*)d_in[5];
  const float* emb2 = (const float*)d_in[6];
  const float* emb3 = (const float*)d_in[7];
  const float* init0_e = (const float*)d_in[8];
  const float* init0   = (const float*)d_in[9];
  const float* W_pre  = (const float*)d_in[10]; const float* b_pre  = (const float*)d_in[11];
  const float* W_msg0 = (const float*)d_in[12]; const float* b_msg0 = (const float*)d_in[13];
  const float* W_q = (const float*)d_in[14]; const float* b_q = (const float*)d_in[15];
  const float* W_k = (const float*)d_in[16]; const float* b_k = (const float*)d_in[17];
  const float* W_v = (const float*)d_in[18]; const float* b_v = (const float*)d_in[19];
  const float* W_r0 = (const float*)d_in[20]; const float* b_r0 = (const float*)d_in[21];
  const float* W_r1 = (const float*)d_in[22]; const float* b_r1 = (const float*)d_in[23];
  const float* W_r2 = (const float*)d_in[24]; const float* b_r2 = (const float*)d_in[25];
  const float* W_post = (const float*)d_in[26]; const float* b_post = (const float*)d_in[27];
  float* out = (float*)d_out;

  char* ws = (char*)d_ws;
  size_t off = 0;
  auto take = [&](size_t bytes) -> void* {
    void* p = ws + off;
    off += (bytes + 255) & ~(size_t)255;
    return p;
  };
  bf16_t* Ph_pre  = (bf16_t*)take((size_t)WIDTH * WIDTH * 2);
  bf16_t* Ph_q    = (bf16_t*)take((size_t)WIDTH * HID * 2);
  bf16_t* Ph_k    = (bf16_t*)take((size_t)WIDTH * HID * 2);
  bf16_t* Ph_v    = (bf16_t*)take((size_t)WIDTH * HID * 2);
  bf16_t* Ph_m    = (bf16_t*)take((size_t)WIDTH * HID * 2);
  bf16_t* Ph_r0   = (bf16_t*)take((size_t)EW * HID * 2);
  bf16_t* Ph_r1   = (bf16_t*)take((size_t)EW * HID * 2);
  bf16_t* Ph_r2   = (bf16_t*)take((size_t)EW * HID * 2);
  bf16_t* Ph_post = (bf16_t*)take((size_t)HID * WIDTH * 2);
  bf16_t* xxh     = (bf16_t*)take((size_t)N_NODES * WIDTH * 2);
  float*  qn  = (float*)take((size_t)N_NODES * HID * 4);
  float*  kn  = (float*)take((size_t)N_NODES * HID * 4);
  float*  vn  = (float*)take((size_t)N_NODES * HID * 4);
  float*  m0  = (float*)take((size_t)N_NODES * HID * 4);
  float*  agg = (float*)take((size_t)N_NODES * HID * 4);

  auto pack = [&](const float* s, bf16_t* d, int K, int Ncols) {
    int n = K * Ncols;
    k_pack_b<<<(n + 255) / 256, 256, 0, stream>>>(s, d, K, Ncols);
  };
  pack(W_pre,  Ph_pre,  WIDTH, WIDTH);
  pack(W_q,    Ph_q,    WIDTH, HID);
  pack(W_k,    Ph_k,    WIDTH, HID);
  pack(W_v,    Ph_v,    WIDTH, HID);
  pack(W_msg0, Ph_m,    WIDTH, HID);
  pack(W_r0,   Ph_r0,   EW, HID);
  pack(W_r1,   Ph_r1,   EW, HID);
  pack(W_r2,   Ph_r2,   EW, HID);
  pack(W_post, Ph_post, HID, WIDTH);

  k_zero<<<(N_NODES * HID + 1023) / 1024, 1024, 0, stream>>>(agg, N_NODES * HID);

  k_pre<<<N_NODES / 16, 512, 0, stream>>>(x, Ph_pre, b_pre, xxh);

  k_node4<<<N_NODES / 16, 256, 0, stream>>>(xxh, Ph_q, Ph_k, Ph_v, Ph_m,
                                            b_q, b_k, b_v, b_msg0,
                                            qn, kn, vn, m0);

  k_edge<<<N_EDGES / 16, 256, 0, stream>>>(edge_index, edge_attr, edge_embed,
                                           emb0, emb1, emb2, emb3, init0_e, init0,
                                           Ph_r0, Ph_r1, Ph_r2, b_r0, b_r1, b_r2,
                                           qn, kn, vn, agg);

  k_final<<<N_NODES / 16, 256, 0, stream>>>(m0, agg, Ph_post, b_post, x, out);
}
